// SD_attn_43344809951780
// MI455X (gfx1250) — matrix operand order repaired and checked by reading, not compiled, not run on hardware
//
#include <hip/hip_runtime.h>
#include <hip/hip_bf16.h>

typedef __attribute__((ext_vector_type(16))) __bf16         v16bf;
typedef __attribute__((ext_vector_type(16))) unsigned short v16u;
typedef __attribute__((ext_vector_type(8)))  unsigned short v8u;
typedef __attribute__((ext_vector_type(4)))  unsigned short v4u;
typedef __attribute__((ext_vector_type(4)))  float          v4f_;
typedef __attribute__((ext_vector_type(8)))  float          v8f;

// ---- helpers ---------------------------------------------------------------

__device__ inline unsigned short f2bf(float f) {
    unsigned u = __builtin_bit_cast(unsigned, f);
    unsigned r = (u + 0x7FFFu + ((u >> 16) & 1u)) >> 16;   // RNE
    return (unsigned short)r;
}

__device__ inline v8f zero8() {
    v8f z;
#pragma unroll
    for (int i = 0; i < 8; ++i) z[i] = 0.f;
    return z;
}

__device__ inline v8u zero8u() {
    v8u z;
#pragma unroll
    for (int i = 0; i < 8; ++i) z[i] = 0;
    return z;
}

// A fragment: 16x32 bf16, row-major A[M][K], ld mult of 8 -> 2x ds_load_b128.
__device__ inline v16bf frag_a(const unsigned short* base, int ld) {
    unsigned lane = threadIdx.x & 31u;
    unsigned m    = lane & 15u;
    unsigned koff = (lane >> 4) << 3;          // 0 or 8
    const unsigned short* row = base + m * ld + koff;
    v8u lo = *(const v8u*)(row);
    v8u hi = *(const v8u*)(row + 16);
    v16u u = __builtin_shufflevector(lo, hi, 0,1,2,3,4,5,6,7,8,9,10,11,12,13,14,15);
    return __builtin_bit_cast(v16bf, u);
}

// B fragment from TRANSPOSED staging Bt[n][k], ld mult of 8 -> 2x ds_load_b128.
__device__ inline v16bf frag_bt(const unsigned short* base, int ld) {
    unsigned lane  = threadIdx.x & 31u;
    unsigned n     = lane & 15u;
    unsigned kbase = (lane >> 4) << 3;         // 0 or 8
    const unsigned short* row = base + n * ld + kbase;
    v8u lo = *(const v8u*)(row);
    v8u hi = *(const v8u*)(row + 16);
    v16u u = __builtin_shufflevector(lo, hi, 0,1,2,3,4,5,6,7,8,9,10,11,12,13,14,15);
    return __builtin_bit_cast(v16bf, u);
}

#define WMMA_BF16(ACC, A, B) \
    (ACC) = __builtin_amdgcn_wmma_f32_16x16x32_bf16(false, (A), false, (B), (short)0, (ACC), false, false)

// ---- constants -------------------------------------------------------------
#define NTOK   147456
#define OUTSZ  56623104      // 2*192*384*384
#define SAVESZ 10616832      // 2048*72*72

// ---- K0: weight prep (fp32 -> bf16, keep [oc][k] layout = transposed-B) ----
__global__ __launch_bounds__(256) void prep_w_kernel(const float* __restrict__ wqkv,
                                                     const float* __restrict__ wproj,
                                                     unsigned short* __restrict__ Wq,
                                                     unsigned short* __restrict__ Wp) {
    int i4 = blockIdx.x * 256 + threadIdx.x;
    if (i4 < 110592) {                         // 1152*384/4
        v4f_ v = *(const v4f_*)(wqkv + (size_t)i4 * 4);
        v4u o;
#pragma unroll
        for (int e = 0; e < 4; ++e) o[e] = f2bf(v[e]);
        *(v4u*)(Wq + (size_t)i4 * 4) = o;
    }
    if (i4 < 36864) {                          // 384*384/4
        v4f_ v = *(const v4f_*)(wproj + (size_t)i4 * 4);
        v4u o;
#pragma unroll
        for (int e = 0; e < 4; ++e) o[e] = f2bf(v[e]);
        *(v4u*)(Wp + (size_t)i4 * 4) = o;
    }
}

// ---- K1: gather (roll + window partition + bf16 convert), 4 elems/thread ---
__global__ __launch_bounds__(256) void gather_kernel(const float* __restrict__ x,
                                                     unsigned short* __restrict__ Xw) {
    int i4 = blockIdx.x * 256 + threadIdx.x;   // NTOK*96 total
    if (i4 >= NTOK * 96) return;
    int t = i4 / 96, ch4 = (i4 % 96) * 4;
    int win = t / 72, n = t % 72;
    int b = win / 1024, rem = win % 1024;
    int wi = rem / 32, wj = rem % 32;
    int r = n / 12, c = n % 12;
    int h = (wi * 6 + r + 3) % 192;
    int w = (wj * 12 + c + 6) % 384;
    size_t src = (((size_t)b * 192 + h) * 384 + w) * 384 + ch4;
    v4f_ v = *(const v4f_*)(x + src);
    v4u o;
#pragma unroll
    for (int e = 0; e < 4; ++e) o[e] = f2bf(v[e]);
    *(v4u*)(Xw + (size_t)t * 384 + ch4) = o;
}

// ---- K2: QKV GEMM (bf16 WMMA) + bias + RoPE + scale, scatter Q/K/V ---------
__global__ __launch_bounds__(256) void qkv_gemm_kernel(const unsigned short* __restrict__ Xw,
                                                       const unsigned short* __restrict__ Wb,
                                                       const float* __restrict__ bqkv,
                                                       unsigned short* __restrict__ Q,
                                                       unsigned short* __restrict__ K,
                                                       unsigned short* __restrict__ V) {
    __shared__ __align__(16) unsigned short As[64 * 40];    // A[m][k], ld 40
    __shared__ __align__(16) unsigned short Bt[128 * 40];   // B^T[n][k], ld 40
    __shared__ float Cs[64 * 128];

    int m0 = blockIdx.x * 64;
    int n0 = blockIdx.y * 128;
    int tid = threadIdx.x;
    int wv  = tid >> 5;
    int mt  = wv >> 1;                  // 0..3 : M tile
    int ntg = (wv & 1) * 4;             // 0 or 4 : N tile group

    v8f acc[4];
#pragma unroll
    for (int j = 0; j < 4; ++j) acc[j] = zero8();

    for (int k0 = 0; k0 < 384; k0 += 32) {
        {   // A: 64x32 = 256 x 8 elems, one b128 per thread
            int r = tid >> 2, c8 = (tid & 3) * 8;
            *(v8u*)(As + r * 40 + c8) = *(const v8u*)(Xw + (size_t)(m0 + r) * 384 + k0 + c8);
        }
#pragma unroll
        for (int it = 0; it < 2; ++it) {   // B: 128x32 = 512 x 8 elems
            int idx = tid + it * 256;
            int c = idx >> 2, r8 = (idx & 3) * 8;
            *(v8u*)(Bt + c * 40 + r8) = *(const v8u*)(Wb + (size_t)(n0 + c) * 384 + k0 + r8);
        }
        __syncthreads();
        v16bf a = frag_a(As + mt * 16 * 40, 40);
#pragma unroll
        for (int j = 0; j < 4; ++j) {
            v16bf b = frag_bt(Bt + (ntg + j) * 16 * 40, 40);
            WMMA_BF16(acc[j], a, b);
        }
        __syncthreads();
    }

    unsigned lane = tid & 31u;
    int nn = lane & 15, mb = (lane >> 4) * 8;
#pragma unroll
    for (int j = 0; j < 4; ++j)
#pragma unroll
        for (int r = 0; r < 8; ++r)
            Cs[(mt * 16 + mb + r) * 128 + (ntg + j) * 16 + nn] = acc[j][r];
    __syncthreads();

    // epilogue: bias + RoPE (channel pairs are adjacent columns of the tile)
    for (int i = tid; i < 64 * 64; i += 256) {
        int r = i >> 6, pc = i & 63;
        int col = pc * 2;
        int t  = m0 + r;
        int oc = n0 + col;
        float v0 = Cs[r * 128 + col]     + bqkv[oc];
        float v1 = Cs[r * 128 + col + 1] + bqkv[oc + 1];
        int part   = oc / 384;             // 0=q 1=k 2=v (uniform per block)
        int within = oc % 384;
        int head   = within >> 6;
        int d0     = within & 63;
        int win = t / 72, n = t % 72;
        size_t dst = (((size_t)win * 6 + head) * 72 + n) * 64 + d0;
        if (part == 2) {
            unsigned pk = (unsigned)f2bf(v0) | ((unsigned)f2bf(v1) << 16);
            *(unsigned*)(V + dst) = pk;
        } else {
            int jj = d0 >> 1;                              // 0..31
            int rr = n / 12, cc = n % 12;
            float inv = __expf(-(float)(jj & 15) * (9.2103403719761836f / 16.f));
            float ang = ((jj < 16) ? (float)rr : (float)cc) * inv;
            float s, c;
            __sincosf(ang, &s, &c);
            float o0 = v0 * c - v1 * s;
            float o1 = v0 * s + v1 * c;
            if (part == 0) { o0 *= 0.125f; o1 *= 0.125f; } // HD^-0.5
            unsigned short* dp = (part == 0) ? Q : K;
            unsigned pk = (unsigned)f2bf(o0) | ((unsigned)f2bf(o1) << 16);
            *(unsigned*)(dp + dst) = pk;
        }
    }
}

// ---- K3: attention, one block per WINDOW, 9 waves, loop over 6 heads -------
// save_attn (mean over heads) accumulated in registers: 288*18 == 72*72.
__global__ __launch_bounds__(288) void attn_kernel(const unsigned short* __restrict__ Q,
                                                   const unsigned short* __restrict__ Kb,
                                                   const unsigned short* __restrict__ Vb,
                                                   unsigned short* __restrict__ O,
                                                   float* __restrict__ save) {
    __shared__ __align__(16) char smem[56832];
    unsigned short* qs  = (unsigned short*)smem;             // A: Q[i][d]    [80][64]
    unsigned short* kbs = (unsigned short*)(smem + 10240);   // B^T: K[j][d]  [80][64]
    unsigned short* Ps  = (unsigned short*)smem;             // A: P[i][j]    [80][96] (alias)
    unsigned short* vst = (unsigned short*)(smem + 20480);   // B^T: V^T[d][j][64][104]
    float*          S   = (float*)(smem + 33792);            // [72][80]

    int win  = blockIdx.x;
    int tid  = threadIdx.x;
    int wave = tid >> 5;
    unsigned lane = tid & 31u;
    int nn = lane & 15, mb = (lane >> 4) * 8;
    int wi = (win % 1024) / 32;
    bool maskrow = (wi == 31);   // only last window-row band has a nonzero mask
    v8u zv = zero8u();

    float sacc[18];
#pragma unroll
    for (int u = 0; u < 18; ++u) sacc[u] = 0.f;

    for (int head = 0; head < 6; ++head) {
        size_t base = ((size_t)win * 6 + head) * 72 * 64;

        // stage Q and K rows (b128 global -> b128 LDS), zero-pad rows 72..79
        for (int idx = tid; idx < 640; idx += 288) {
            int r = idx >> 3, c8 = (idx & 7) * 8;
            v8u q8 = zv, k8 = zv;
            if (r < 72) {
                q8 = *(const v8u*)(Q  + base + r * 64 + c8);
                k8 = *(const v8u*)(Kb + base + r * 64 + c8);
            }
            *(v8u*)(qs  + r * 64 + c8) = q8;
            *(v8u*)(kbs + r * 64 + c8) = k8;
        }
        // stage V transposed: vst[d][j], cols 72..103 zero
        for (int idx = tid; idx < 832; idx += 288) {
            int j = idx >> 3, db = (idx & 7) * 8;
            v8u v8 = zv;
            if (j < 72) v8 = *(const v8u*)(Vb + base + j * 64 + db);
#pragma unroll
            for (int e = 0; e < 8; ++e) vst[(db + e) * 104 + j] = v8[e];
        }
        __syncthreads();

        // S = Q @ K^T : 5x5 tiles, K=64 (2 steps)
        for (int t = wave; t < 25; t += 9) {
            int mt = t / 5, nt = t % 5;
            v8f acc = zero8();
#pragma unroll
            for (int ks = 0; ks < 2; ++ks) {
                v16bf a = frag_a(qs + mt * 16 * 64 + ks * 32, 64);
                v16bf b = frag_bt(kbs + nt * 16 * 64 + ks * 32, 64);
                WMMA_BF16(acc, a, b);
            }
#pragma unroll
            for (int r = 0; r < 8; ++r) {
                int i = mt * 16 + mb + r;
                if (i < 72) S[i * 80 + nt * 16 + nn] = acc[r];
            }
        }
        __syncthreads();

        for (int idx = tid; idx < 960; idx += 288) *(v8u*)(Ps + idx * 8) = zv;
        __syncthreads();

        // softmax per row + on-the-fly mask; write normalized f32 back + bf16 P
        if (tid < 72) {
            int i = tid;
            bool gi = (i / 12) < 3;
            float mx = -1e30f;
            for (int j = 0; j < 72; ++j) {
                bool msk = maskrow && (((j / 12) < 3) != gi);
                float s = S[i * 80 + j];
                if (!msk && s > mx) mx = s;
            }
            float sum = 0.f;
            for (int j = 0; j < 72; ++j) {
                bool msk = maskrow && (((j / 12) < 3) != gi);
                float p = msk ? 0.f : __expf(S[i * 80 + j] - mx);
                sum += p;
                S[i * 80 + j] = p;
            }
            float inv = 1.f / sum;
            for (int j = 0; j < 72; ++j) {
                float p = S[i * 80 + j] * inv;
                S[i * 80 + j] = p;
                Ps[i * 96 + j] = f2bf(p);
            }
        }
        __syncthreads();

        // accumulate head-mean into registers (18 elems per thread)
        {
            int e0 = tid * 18;
#pragma unroll
            for (int u = 0; u < 18; ++u) {
                int e = e0 + u;
                sacc[u] += S[(e / 72) * 80 + (e % 72)];
            }
        }

        // O = P @ V : 5x4 tiles, K=96 (3 steps; cols 72..95 zero)
        for (int t = wave; t < 20; t += 9) {
            int mt = t / 4, nt = t % 4;
            v8f acc = zero8();
#pragma unroll
            for (int ks = 0; ks < 3; ++ks) {
                v16bf a = frag_a(Ps + mt * 16 * 96 + ks * 32, 96);
                v16bf b = frag_bt(vst + nt * 16 * 104 + ks * 32, 104);
                WMMA_BF16(acc, a, b);
            }
#pragma unroll
            for (int r = 0; r < 8; ++r) {
                int i = mt * 16 + mb + r;
                if (i < 72)
                    O[((size_t)win * 72 + i) * 384 + head * 64 + nt * 16 + nn] = f2bf(acc[r]);
            }
        }
        __syncthreads();   // protect qs/Ps/vst/S before next head
    }

    // write save_attn: contiguous 18 floats per thread, no atomics
    {
        size_t b = (size_t)win * 5184 + (size_t)tid * 18;
#pragma unroll
        for (int u = 0; u < 18; ++u) save[b + u] = sacc[u] * (1.f / 6.f);
    }
}

// ---- K4: output projection + window-reverse + roll back --------------------
__global__ __launch_bounds__(256) void proj_gemm_kernel(const unsigned short* __restrict__ O,
                                                        const unsigned short* __restrict__ Wp,
                                                        const float* __restrict__ bproj,
                                                        float* __restrict__ out) {
    __shared__ __align__(16) unsigned short As[64 * 40];
    __shared__ __align__(16) unsigned short Bt[128 * 40];
    __shared__ float Cs[64 * 128];

    int m0 = blockIdx.x * 64;
    int n0 = blockIdx.y * 128;
    int tid = threadIdx.x;
    int wv  = tid >> 5;
    int mt  = wv >> 1;
    int ntg = (wv & 1) * 4;

    v8f acc[4];
#pragma unroll
    for (int j = 0; j < 4; ++j) acc[j] = zero8();

    for (int k0 = 0; k0 < 384; k0 += 32) {
        {
            int r = tid >> 2, c8 = (tid & 3) * 8;
            *(v8u*)(As + r * 40 + c8) = *(const v8u*)(O + (size_t)(m0 + r) * 384 + k0 + c8);
        }
#pragma unroll
        for (int it = 0; it < 2; ++it) {
            int idx = tid + it * 256;
            int c = idx >> 2, r8 = (idx & 3) * 8;
            *(v8u*)(Bt + c * 40 + r8) = *(const v8u*)(Wp + (size_t)(n0 + c) * 384 + k0 + r8);
        }
        __syncthreads();
        v16bf a = frag_a(As + mt * 16 * 40, 40);
#pragma unroll
        for (int j = 0; j < 4; ++j) {
            v16bf b = frag_bt(Bt + (ntg + j) * 16 * 40, 40);
            WMMA_BF16(acc[j], a, b);
        }
        __syncthreads();
    }

    unsigned lane = tid & 31u;
    int nn = lane & 15, mb = (lane >> 4) * 8;
#pragma unroll
    for (int j = 0; j < 4; ++j)
#pragma unroll
        for (int r = 0; r < 8; ++r)
            Cs[(mt * 16 + mb + r) * 128 + (ntg + j) * 16 + nn] = acc[j][r];
    __syncthreads();

    for (int i = tid; i < 64 * 128; i += 256) {
        int r = i >> 7, c = i & 127;
        int t  = m0 + r;
        int oc = n0 + c;
        float val = Cs[r * 128 + c] + bproj[oc];
        int win = t / 72, n = t % 72;
        int b = win / 1024, rem = win % 1024;
        int wi = rem / 32, wj = rem % 32;
        int rr = n / 12, cc = n % 12;
        int h = (wi * 6 + rr + 3) % 192;
        int w = (wj * 12 + cc + 6) % 384;
        out[(((size_t)b * 192 + h) * 384 + w) * 384 + oc] = val;
    }
}

// ---- launcher --------------------------------------------------------------
extern "C" void kernel_launch(void* const* d_in, const int* in_sizes, int n_in,
                              void* d_out, int out_size, void* d_ws, size_t ws_size,
                              hipStream_t stream) {
    const float* x      = (const float*)d_in[0];
    const float* w_qkv  = (const float*)d_in[1];
    const float* b_qkv  = (const float*)d_in[2];
    const float* w_proj = (const float*)d_in[3];
    const float* b_proj = (const float*)d_in[4];
    float* out  = (float*)d_out;
    float* save = out + OUTSZ;

    char* ws = (char*)d_ws;
    const size_t SZ = (size_t)NTOK * 384 * 2;     // 113,246,208 bytes
    unsigned short* Xw = (unsigned short*)(ws);
    unsigned short* Qb = (unsigned short*)(ws + SZ);
    unsigned short* Kb = (unsigned short*)(ws + 2 * SZ);
    unsigned short* Vb = (unsigned short*)(ws + 3 * SZ);
    unsigned short* Wq = (unsigned short*)(ws + 4 * SZ);
    unsigned short* Wp = (unsigned short*)(ws + 4 * SZ + (size_t)1152 * 384 * 2);
    unsigned short* Ob = Xw;                      // reuse: Xw dead after QKV GEMM

    prep_w_kernel<<<432, 256, 0, stream>>>(w_qkv, w_proj, Wq, Wp);
    gather_kernel<<<(NTOK * 96) / 256, 256, 0, stream>>>(x, Xw);
    qkv_gemm_kernel<<<dim3(NTOK / 64, 9), 256, 0, stream>>>(Xw, Wq, b_qkv, Qb, Kb, Vb);
    attn_kernel<<<2048, 288, 0, stream>>>(Qb, Kb, Vb, Ob, save);
    proj_gemm_kernel<<<dim3(NTOK / 64, 3), 256, 0, stream>>>(Ob, Wp, b_proj, out);
}
